// GraphSAGE_49057116455172
// MI455X (gfx1250) — matrix operand order repaired and checked by reading, not compiled, not run on hardware
//
#include <hip/hip_runtime.h>
#include <hip/hip_bf16.h>

// ---------------------------------------------------------------------------
// GraphSAGE 2-layer inference for MI455X (gfx1250, wave32).
//   msg_i = sum_{(s->i)} feat[s]   (f32 global atomics, L2-resident: all node
//                                   matrices are 51.2 MB vs 192 MB L2)
//   agg   = msg / max(deg,1)
//   h     = relu(agg@W1_l + x@W1_r + b1)      (bf16 WMMA, f32 accum)
//   out   = agg2@W2_l + h@W2_r + b2           (bf16 WMMA, f32 accum)
//   log_softmax over cols [0,6) and [6,14)
// ---------------------------------------------------------------------------

typedef __attribute__((ext_vector_type(16))) __bf16 v16bf;
typedef __attribute__((ext_vector_type(8)))  float  v8f;

union V16 {
    v16bf v;
    unsigned short s[16];
    unsigned int   u[8];
    uint4          q[2];
};

// Native f32 -> bf16 (RNE) so the backend can fuse v_cvt_pk_bf16_f32.
__device__ __forceinline__ unsigned short bf_bits(float f) {
    return __builtin_bit_cast(unsigned short, (__bf16)f);
}

// Build the 16x32 bf16 A-operand chunk for one lane.
// Lane layout (ISA 7.12.2, 16-bit A 16x32): lanes 0-15 row M=lane hold
// K = kbase+{0..7} and kbase+{16..23}; lanes 16-31 row M=lane-16 hold
// K = kbase+{8..15} and kbase+{24..31}.  hh = lane>>4.
__device__ __forceinline__ V16 make_a_chunk(const float* __restrict__ rowp,
                                            int kbase, int hh, float scale) {
    const float* p0 = rowp + kbase + 8 * hh;
    const float* p1 = rowp + kbase + 16 + 8 * hh;
    float4 a0 = *(const float4*)(p0);
    float4 a1 = *(const float4*)(p0 + 4);
    float4 b0 = *(const float4*)(p1);
    float4 b1 = *(const float4*)(p1 + 4);
    V16 r;
    r.s[0]  = bf_bits(a0.x * scale); r.s[1]  = bf_bits(a0.y * scale);
    r.s[2]  = bf_bits(a0.z * scale); r.s[3]  = bf_bits(a0.w * scale);
    r.s[4]  = bf_bits(a1.x * scale); r.s[5]  = bf_bits(a1.y * scale);
    r.s[6]  = bf_bits(a1.z * scale); r.s[7]  = bf_bits(a1.w * scale);
    r.s[8]  = bf_bits(b0.x * scale); r.s[9]  = bf_bits(b0.y * scale);
    r.s[10] = bf_bits(b0.z * scale); r.s[11] = bf_bits(b0.w * scale);
    r.s[12] = bf_bits(b1.x * scale); r.s[13] = bf_bits(b1.y * scale);
    r.s[14] = bf_bits(b1.z * scale); r.s[15] = bf_bits(b1.w * scale);
    return r;
}

// Load a pre-packed 32x16 bf16 B-operand chunk (32 contiguous bytes / lane).
__device__ __forceinline__ v16bf load_b_chunk(const unsigned short* __restrict__ packed,
                                              int s, int t, int ntiles, int lane) {
    const uint4* p = (const uint4*)(packed + ((size_t)((s * ntiles + t) * 32 + lane)) * 16);
    V16 r;
    r.q[0] = p[0];
    r.q[1] = p[1];
    return r.v;
}

// --------------------------- weight pre-pack -------------------------------
// B lane layout (dense 16-bit B, K x 16): lane = column (lane&15);
// lanes 0-15 hold K = kbase+{0..7,16..23}, lanes 16-31 hold K = kbase+{8..15,24..31}.
__global__ void pack_w_kernel(const float* __restrict__ W, unsigned short* __restrict__ out,
                              int ncols_true, int ntiles) {
    int idx = blockIdx.x * blockDim.x + threadIdx.x;
    int total = 4 * ntiles * 32 * 16;
    if (idx >= total) return;
    int j    = idx & 15;
    int lane = (idx >> 4) & 31;
    int t    = (idx >> 9) % ntiles;
    int s    = idx / (512 * ntiles);
    int n  = lane & 15;
    int hh = lane >> 4;
    int k   = s * 32 + (j < 8 ? hh * 8 + j : 16 + hh * 8 + (j - 8));
    int col = t * 16 + n;
    float val = (col < ncols_true) ? W[(size_t)k * ncols_true + col] : 0.0f;
    out[idx] = bf_bits(val);
}

// ------------------------------ degree -------------------------------------
__global__ void deg_kernel(const long long* __restrict__ dst, float* __restrict__ deg, int E) {
    int e = blockIdx.x * blockDim.x + threadIdx.x;
    if (e >= E) return;
    unsafeAtomicAdd(&deg[(int)dst[e]], 1.0f);
}

// --------------------------- edge scatter-add ------------------------------
// One wave (32 lanes) per edge; each lane moves 4 contiguous floats.
// 512 B contiguous gather per edge + f32 atomic adds that resolve in L2.
__global__ void scatter_kernel(const float* __restrict__ feat,
                               const long long* __restrict__ src,
                               const long long* __restrict__ dst,
                               float* __restrict__ msg, int E) {
    int t = blockIdx.x * blockDim.x + threadIdx.x;
    int edge = t >> 5;
    if (edge >= E) return;
    int lane = t & 31;
    int s = (int)src[edge];
    int d = (int)dst[edge];
    float4 v = *(const float4*)(feat + (size_t)s * 128 + lane * 4);
    float* out = msg + (size_t)d * 128 + lane * 4;
    unsafeAtomicAdd(out + 0, v.x);
    unsafeAtomicAdd(out + 1, v.y);
    unsafeAtomicAdd(out + 2, v.z);
    unsafeAtomicAdd(out + 3, v.w);
}

// ------------------------------- layer 1 -----------------------------------
// Block = 256 threads = 8 waves, one 16-row tile per block, wave w handles
// output columns [16w, 16w+16).  The agg/x tile is converted to bf16 ONCE per
// block into LDS in exact WMMA A-operand order (2 mats x 4 ksteps x 32 lanes
// x 32 B = 8 KB of the 320 KB WGP LDS); thread <-> (mat,kstep,lane) is a
// perfect 256-way split.  Each wave then reads its per-lane 32 B chunks with
// ds_load_b128 and issues 8 x v_wmma_f32_16x16x32_bf16.
__global__ void gemm1_kernel(const float* __restrict__ msg, const float* __restrict__ deg,
                             const float* __restrict__ x,
                             const unsigned short* __restrict__ Bl,
                             const unsigned short* __restrict__ Br,
                             const float* __restrict__ bias,
                             float* __restrict__ h) {
    __shared__ __align__(32) unsigned short ldsA[2][4][32][16];

    int tid = threadIdx.x;
    {
        int m    = tid >> 7;         // 0 = agg (msg/deg), 1 = x
        int s    = (tid >> 5) & 3;   // kstep
        int lane = tid & 31;
        int r  = lane & 15;
        int hh = lane >> 4;
        int row = blockIdx.x * 16 + r;
        const float* base = (m ? x : msg) + (size_t)row * 128;
        float scale = m ? 1.0f : (1.0f / fmaxf(deg[row], 1.0f));
        V16 c = make_a_chunk(base, s * 32, hh, scale);
        uint4* o = (uint4*)&ldsA[m][s][lane][0];
        o[0] = c.q[0];
        o[1] = c.q[1];
    }
    __syncthreads();

    int lane    = tid & 31;
    int colTile = tid >> 5;          // 0..7  (H = 128)
    v8f c = {0.f, 0.f, 0.f, 0.f, 0.f, 0.f, 0.f, 0.f};
#pragma unroll
    for (int s = 0; s < 4; ++s) {
        V16 aA, aX;
        const uint4* qa = (const uint4*)&ldsA[0][s][lane][0];
        aA.q[0] = qa[0]; aA.q[1] = qa[1];
        const uint4* qx = (const uint4*)&ldsA[1][s][lane][0];
        aX.q[0] = qx[0]; aX.q[1] = qx[1];
        v16bf bL = load_b_chunk(Bl, s, colTile, 8, lane);
        c = __builtin_amdgcn_wmma_f32_16x16x32_bf16(false, aA.v, false, bL, (short)0, c, false, false);
        v16bf bR = load_b_chunk(Br, s, colTile, 8, lane);
        c = __builtin_amdgcn_wmma_f32_16x16x32_bf16(false, aX.v, false, bR, (short)0, c, false, false);
    }

    int n  = lane & 15;
    int hh = lane >> 4;
    float bv = bias[colTile * 16 + n];
#pragma unroll
    for (int i = 0; i < 8; ++i) {
        int m = blockIdx.x * 16 + i + 8 * hh;        // C layout: VGPR i -> row i (+8 hi half)
        float v = fmaxf(c[i] + bv, 0.0f);
        h[(size_t)m * 128 + colTile * 16 + n] = v;
    }
}

// ------------------------------- layer 2 -----------------------------------
// C = 14 -> one padded 16-column tile; wave w of each block takes row tile
// blockIdx.x*8 + w (distinct tiles -> no cross-wave sharing; direct path).
__global__ void gemm2_kernel(const float* __restrict__ msg, const float* __restrict__ deg,
                             const float* __restrict__ h,
                             const unsigned short* __restrict__ Bl,
                             const unsigned short* __restrict__ Br,
                             const float* __restrict__ bias,
                             float* __restrict__ out_raw, int rowTiles) {
    int lane = threadIdx.x & 31;
    int wave = threadIdx.x >> 5;
    int rowTile = blockIdx.x * 8 + wave;
    if (rowTile >= rowTiles) return;                 // uniform per wave -> EXEC stays all-ones
    int hh = lane >> 4;
    int r  = lane & 15;
    int row = rowTile * 16 + r;
    float rdeg = 1.0f / fmaxf(deg[row], 1.0f);
    const float* mrow = msg + (size_t)row * 128;
    const float* hrow = h   + (size_t)row * 128;

    v8f c = {0.f, 0.f, 0.f, 0.f, 0.f, 0.f, 0.f, 0.f};
#pragma unroll
    for (int s = 0; s < 4; ++s) {
        int kbase = s * 32;
        V16 aA = make_a_chunk(mrow, kbase, hh, rdeg);
        v16bf bL = load_b_chunk(Bl, s, 0, 1, lane);
        c = __builtin_amdgcn_wmma_f32_16x16x32_bf16(false, aA.v, false, bL, (short)0, c, false, false);
        V16 aH = make_a_chunk(hrow, kbase, hh, 1.0f);
        v16bf bR = load_b_chunk(Br, s, 0, 1, lane);
        c = __builtin_amdgcn_wmma_f32_16x16x32_bf16(false, aH.v, false, bR, (short)0, c, false, false);
    }

    int n = lane & 15;
    float bv = (n < 14) ? bias[n] : 0.0f;
#pragma unroll
    for (int i = 0; i < 8; ++i) {
        int m = rowTile * 16 + i + 8 * hh;
        out_raw[(size_t)m * 16 + n] = c[i] + bv;
    }
}

// ----------------------------- log-softmax ---------------------------------
__global__ void lsm_kernel(const float* __restrict__ out_raw, float* __restrict__ out, int N) {
    int i = blockIdx.x * blockDim.x + threadIdx.x;
    if (i >= N) return;
    const float* rp = out_raw + (size_t)i * 16;
    float v[14];
#pragma unroll
    for (int j = 0; j < 14; ++j) v[j] = rp[j];

    float m1 = v[0];
#pragma unroll
    for (int j = 1; j < 6; ++j) m1 = fmaxf(m1, v[j]);
    float s1 = 0.0f;
#pragma unroll
    for (int j = 0; j < 6; ++j) s1 += __expf(v[j] - m1);
    float l1 = m1 + __logf(s1);
#pragma unroll
    for (int j = 0; j < 6; ++j) out[(size_t)i * 6 + j] = v[j] - l1;

    float m2 = v[6];
#pragma unroll
    for (int j = 7; j < 14; ++j) m2 = fmaxf(m2, v[j]);
    float s2 = 0.0f;
#pragma unroll
    for (int j = 0; j < 8; ++j) s2 += __expf(v[6 + j] - m2);
    float l2 = m2 + __logf(s2);
    float* o2 = out + (size_t)N * 6;
#pragma unroll
    for (int j = 0; j < 8; ++j) o2[(size_t)i * 8 + j] = v[6 + j] - l2;
}

// ------------------------------- launcher ----------------------------------
extern "C" void kernel_launch(void* const* d_in, const int* in_sizes, int n_in,
                              void* d_out, int out_size, void* d_ws, size_t ws_size,
                              hipStream_t stream) {
    const float*      x   = (const float*)d_in[0];
    const long long*  ei  = (const long long*)d_in[1];
    const float*      W1l = (const float*)d_in[3];
    const float*      W1r = (const float*)d_in[4];
    const float*      b1  = (const float*)d_in[5];
    const float*      W2l = (const float*)d_in[6];
    const float*      W2r = (const float*)d_in[7];
    const float*      b2  = (const float*)d_in[8];

    const int N = in_sizes[0] / 128;   // 100000
    const int E = in_sizes[1] / 2;     // 1600000
    const long long* src = ei;
    const long long* dst = ei + E;

    char* ws = (char*)d_ws;
    size_t off = 0;
    auto alloc = [&](size_t bytes) -> void* {
        void* p = ws + off;
        off = (off + bytes + 255) & ~(size_t)255;
        return p;
    };
    float*          msg     = (float*)alloc((size_t)N * 128 * 4);
    float*          deg     = (float*)alloc((size_t)N * 4);
    float*          h       = (float*)alloc((size_t)N * 128 * 4);
    float*          out_raw = (float*)alloc((size_t)N * 16 * 4);
    unsigned short* B1l     = (unsigned short*)alloc((size_t)4 * 8 * 32 * 16 * 2);
    unsigned short* B1r     = (unsigned short*)alloc((size_t)4 * 8 * 32 * 16 * 2);
    unsigned short* B2l     = (unsigned short*)alloc((size_t)4 * 1 * 32 * 16 * 2);
    unsigned short* B2r     = (unsigned short*)alloc((size_t)4 * 1 * 32 * 16 * 2);

    hipMemsetAsync(msg, 0, (size_t)N * 128 * 4, stream);
    hipMemsetAsync(deg, 0, (size_t)N * 4, stream);

    pack_w_kernel<<<(4 * 8 * 512 + 255) / 256, 256, 0, stream>>>(W1l, B1l, 128, 8);
    pack_w_kernel<<<(4 * 8 * 512 + 255) / 256, 256, 0, stream>>>(W1r, B1r, 128, 8);
    pack_w_kernel<<<(4 * 1 * 512 + 255) / 256, 256, 0, stream>>>(W2l, B2l, 14, 1);
    pack_w_kernel<<<(4 * 1 * 512 + 255) / 256, 256, 0, stream>>>(W2r, B2r, 14, 1);

    deg_kernel<<<(E + 255) / 256, 256, 0, stream>>>(dst, deg, E);

    // Layer 1
    scatter_kernel<<<(int)(((size_t)E * 32 + 255) / 256), 256, 0, stream>>>(x, src, dst, msg, E);
    gemm1_kernel<<<N / 16, 256, 0, stream>>>(msg, deg, x, B1l, B1r, b1, h);

    // Layer 2 (same graph -> same deg; just reset msg)
    hipMemsetAsync(msg, 0, (size_t)N * 128 * 4, stream);
    scatter_kernel<<<(int)(((size_t)E * 32 + 255) / 256), 256, 0, stream>>>(h, src, dst, msg, E);
    int rowTiles = N / 16;
    gemm2_kernel<<<(rowTiles + 7) / 8, 256, 0, stream>>>(msg, deg, h, B2l, B2r, b2, out_raw, rowTiles);

    lsm_kernel<<<(N + 255) / 256, 256, 0, stream>>>(out_raw, (float*)d_out, N);
}
